// GraphAttentionEncoder_67903432949848
// MI455X (gfx1250) — compile-verified, matrix operand order checked by reading, not run on hardware
//
#include <hip/hip_runtime.h>
#include <hip/hip_bf16.h>
#include <math.h>

// ---------------- problem constants (match reference) ----------------
#define NB    32          // graphs
#define NPG0  2048        // nodes per graph
#define NTOT  (NB*NPG0)   // 65536
#define EPG   16384
#define ETOT  (NB*EPG)    // 524288
#define FIN   129
#define KPAD0 160         // FIN zero-padded to multiple of 32
#define HC    256
#define NHEAD 4
#define CH    64

typedef __attribute__((ext_vector_type(16))) __bf16 v16bf;
typedef __attribute__((ext_vector_type(8)))  __bf16 v8bf;
typedef __attribute__((ext_vector_type(8)))  float  v8f;

// ---------------- small helpers ----------------
__device__ inline void atomicMaxF(float* addr, float v) {
  if (v >= 0.f) atomicMax((int*)addr, __float_as_int(v));
  else          atomicMin((unsigned int*)addr, __float_as_uint(v));
}

__global__ void k_fill_f32(float* __restrict__ p, float v, int n) {
  int i = blockIdx.x * blockDim.x + threadIdx.x;
  if (i < n) p[i] = v;
}
__global__ void k_fill_i32(int* __restrict__ p, int v, int n) {
  int i = blockIdx.x * blockDim.x + threadIdx.x;
  if (i < n) p[i] = v;
}

// ---------------- f32 -> bf16 packing (A: row-major, zero-padded K) ----------------
// grid.x = M rows, blockDim.x = Kp
__global__ void k_cvt_a_bf16(const float* __restrict__ in, __bf16* __restrict__ out,
                             int Kin, int Kp) {
  const int r = blockIdx.x, k = threadIdx.x;
  out[(size_t)r * Kp + k] = (__bf16)((k < Kin) ? in[(size_t)r * Kin + k] : 0.f);
}

// ---------------- f32 -> bf16 packing (B: [Kin,256] -> column-major [256,Kp]) ----------------
// grid.x = 256 cols, blockDim.x = Kp
__global__ void k_cvt_b_bf16(const float* __restrict__ in, __bf16* __restrict__ out,
                             int Kin, int Kp) {
  const int col = blockIdx.x, k = threadIdx.x;
  out[(size_t)col * Kp + k] = (__bf16)((k < Kin) ? in[(size_t)k * HC + col] : 0.f);
}

// ---------------- bf16 WMMA GEMM:  out[M,256] = act(A[M,Kp] @ Bt[256,Kp]^T + bias) ----------------
// Block = 256 threads = 8 waves. Each block: 16 rows. Each wave: 2 column tiles of 16.
// A-operand lane layout (16x32 bf16): lane<16 row=lane k={0..7,16..23}; lane>=16 k={8..15,24..31}.
// B-operand mirrors with lane = column; Bt is column-major so both are 2x contiguous b128 loads.
__global__ __launch_bounds__(256)
void k_gemm_bf16(const __bf16* __restrict__ A, const __bf16* __restrict__ Bt,
                 const float* __restrict__ bias, float* __restrict__ out,
                 int Kp, int act) {
  const int lane = threadIdx.x & 31;
  const int wave = threadIdx.x >> 5;
  const int rowbase = blockIdx.x << 4;
  const int rlo = lane & 15;
  const int hi  = lane >> 4;             // 0 or 1
  const int row = rowbase + rlo;
  const int col0 = (wave * 2) << 4;
  const int col1 = col0 + 16;
  const v8bf* __restrict__ ap  = (const v8bf*)(A  + (size_t)row * Kp);
  const v8bf* __restrict__ b0p = (const v8bf*)(Bt + (size_t)(col0 + rlo) * Kp);
  const v8bf* __restrict__ b1p = (const v8bf*)(Bt + (size_t)(col1 + rlo) * Kp);
  v8f c0 = {}; v8f c1 = {};
  const int nq = Kp >> 3;                 // #v8bf chunks per row
  for (int q = hi; q < nq; q += 4) {      // q = k0/8 + hi, k0 += 32
    v8bf alo  = ap[q],  ahi  = ap[q + 2];
    v8bf b0lo = b0p[q], b0hi = b0p[q + 2];
    v8bf b1lo = b1p[q], b1hi = b1p[q + 2];
    v16bf a  = __builtin_shufflevector(alo,  ahi,  0,1,2,3,4,5,6,7,8,9,10,11,12,13,14,15);
    v16bf b0 = __builtin_shufflevector(b0lo, b0hi, 0,1,2,3,4,5,6,7,8,9,10,11,12,13,14,15);
    v16bf b1 = __builtin_shufflevector(b1lo, b1hi, 0,1,2,3,4,5,6,7,8,9,10,11,12,13,14,15);
    c0 = __builtin_amdgcn_wmma_f32_16x16x32_bf16(false, a, false, b0, (short)0, c0, false, false);
    c1 = __builtin_amdgcn_wmma_f32_16x16x32_bf16(false, a, false, b1, (short)0, c1, false, false);
  }
#pragma unroll
  for (int r = 0; r < 8; ++r) {
    const int m = rowbase + r + hi * 8;
    float v0 = c0[r], v1 = c1[r];
    if (bias) { v0 += bias[col0 + rlo]; v1 += bias[col1 + rlo]; }
    if (act == 1) { v0 = fmaxf(v0, 0.f); v1 = fmaxf(v1, 0.f); }
    out[(size_t)m * HC + col0 + rlo] = v0;
    out[(size_t)m * HC + col1 + rlo] = v1;
  }
}

// ---------------- attention per-node dots: as[n,h]=<g[n,h,:],asrc[h,:]>, ad likewise ----------------
__global__ void k_att(const float* __restrict__ g, const float* __restrict__ asrc,
                      const float* __restrict__ adst, float* __restrict__ as_,
                      float* __restrict__ ad_, int Nn) {
  int tid = blockIdx.x * blockDim.x + threadIdx.x;
  if (tid >= Nn * NHEAD) return;
  int n = tid >> 2, h = tid & 3;
  const float4* gp = (const float4*)(g + (size_t)n * HC + h * CH);
  const float4* sp = (const float4*)(asrc + h * CH);
  const float4* dp = (const float4*)(adst + h * CH);
  float a1 = 0.f, a2 = 0.f;
#pragma unroll 4
  for (int q = 0; q < 16; ++q) {
    float4 v = gp[q], w1 = sp[q], w2 = dp[q];
    a1 += v.x*w1.x + v.y*w1.y + v.z*w1.z + v.w*w1.w;
    a2 += v.x*w2.x + v.y*w2.y + v.z*w2.z + v.w*w2.w;
  }
  as_[n * NHEAD + h] = a1;
  ad_[n * NHEAD + h] = a2;
}

// ---------------- segment softmax over destinations (3 passes + self loops) ----------------
__global__ void k_edge_logit(const int* __restrict__ src, const int* __restrict__ dst,
                             const float* __restrict__ as_, const float* __restrict__ ad_,
                             float* __restrict__ logit, float* __restrict__ mx, int E, int Nn) {
  int e = blockIdx.x * blockDim.x + threadIdx.x;
  if (e >= E + Nn) return;
  int s, d;
  if (e < E) { s = src[e]; d = dst[e]; if (s < 0 || d < 0) return; }
  else       { s = d = e - E; }
#pragma unroll
  for (int h = 0; h < NHEAD; ++h) {
    float z = as_[s * NHEAD + h] + ad_[d * NHEAD + h];
    z = (z > 0.f) ? z : 0.2f * z;            // LeakyReLU(0.2)
    logit[(size_t)e * NHEAD + h] = z;
    atomicMaxF(&mx[d * NHEAD + h], z);
  }
}

__global__ void k_edge_expsum(const int* __restrict__ dst, const float* __restrict__ mx,
                              float* __restrict__ logit, float* __restrict__ den, int E, int Nn) {
  int e = blockIdx.x * blockDim.x + threadIdx.x;
  if (e >= E + Nn) return;
  int d;
  if (e < E) { d = dst[e]; if (d < 0) return; }
  else       { d = e - E; }
#pragma unroll
  for (int h = 0; h < NHEAD; ++h) {
    float z = logit[(size_t)e * NHEAD + h];
    float ex = __expf(z - mx[d * NHEAD + h]);
    logit[(size_t)e * NHEAD + h] = ex;
    atomicAdd(&den[d * NHEAD + h], ex);
  }
}

__global__ void k_edge_scatter(const int* __restrict__ src, const int* __restrict__ dst,
                               const float* __restrict__ ex_, const float* __restrict__ den,
                               const float* __restrict__ g, float* __restrict__ acc, int E, int Nn) {
  int tid = blockIdx.x * blockDim.x + threadIdx.x;
  if (tid >= (E + Nn) * NHEAD) return;
  int e = tid >> 2, h = tid & 3;
  int s, d;
  if (e < E) { s = src[e]; d = dst[e]; if (s < 0 || d < 0) return; }
  else       { s = d = e - E; }
  float coef = ex_[(size_t)e * NHEAD + h] / (den[d * NHEAD + h] + 1e-16f);
  const float4* gs = (const float4*)(g + (size_t)s * HC + h * CH);
  float* od = acc + (size_t)d * HC + h * CH;
#pragma unroll 4
  for (int q = 0; q < 16; ++q) {
    float4 v = gs[q]; int c = q * 4;
    atomicAdd(&od[c + 0], v.x * coef);
    atomicAdd(&od[c + 1], v.y * coef);
    atomicAdd(&od[c + 2], v.z * coef);
    atomicAdd(&od[c + 3], v.w * coef);
  }
}

__global__ void k_bias_elu(float* __restrict__ p, const float* __restrict__ b, int count) {
  int i = blockIdx.x * blockDim.x + threadIdx.x;
  if (i >= count) return;
  float v = p[i] + b[i & (HC - 1)];
  p[i] = (v > 0.f) ? v : (__expf(v) - 1.f);
}

// ---------------- TopK pooling ----------------
__global__ void k_score(const float* __restrict__ h, const float* __restrict__ w,
                        float* __restrict__ score, int Nn) {
  int n = blockIdx.x * blockDim.x + threadIdx.x;
  if (n >= Nn) return;
  const float4* hp = (const float4*)(h + (size_t)n * HC);
  const float4* wp = (const float4*)w;
  float dp = 0.f, wn = 0.f;
  for (int q = 0; q < HC / 4; ++q) {
    float4 a = hp[q], b = wp[q];
    dp += a.x*b.x + a.y*b.y + a.z*b.z + a.w*b.w;
    wn += b.x*b.x + b.y*b.y + b.z*b.z + b.w*b.w;
  }
  score[n] = dp / (sqrtf(wn) + 1e-16f);
}

// block per graph; blockDim = n_in/2 = k; LDS bitonic sort (descending)
__global__ void k_topk(const float* __restrict__ score, float* __restrict__ tfac,
                       int* __restrict__ perm, int* __restrict__ newid, int n_in) {
  __shared__ float ss[2048];
  __shared__ int   si[2048];
  const int g = blockIdx.x, t = threadIdx.x;
  const int k = n_in >> 1;                  // == blockDim.x
  const int base = g * n_in;
  for (int i = t; i < n_in; i += k) { ss[i] = score[base + i]; si[i] = i; }
  __syncthreads();
  for (int ksz = 2; ksz <= n_in; ksz <<= 1)
    for (int j = ksz >> 1; j > 0; j >>= 1) {
      int i = ((t & ~(j - 1)) << 1) | (t & (j - 1));
      int l = i | j;
      bool desc = ((i & ksz) == 0);
      float a = ss[i], b = ss[l];
      if ((a < b) == desc) {
        ss[i] = b; ss[l] = a;
        int tmp = si[i]; si[i] = si[l]; si[l] = tmp;
      }
      __syncthreads();
    }
  int gnode = base + si[t];
  int nidx  = g * k + t;
  perm[nidx] = gnode;
  tfac[nidx] = tanhf(ss[t]);
  newid[gnode] = nidx;
}

__global__ void k_pool_gather(const float* __restrict__ hin, const int* __restrict__ perm,
                              const float* __restrict__ tfac, float* __restrict__ hout, int nout) {
  int tid = blockIdx.x * blockDim.x + threadIdx.x;   // nout*64 float4 slots
  if (tid >= nout * (HC / 4)) return;
  int i = tid >> 6, q = tid & 63;
  int old = perm[i]; float f = tfac[i];
  float4 v = ((const float4*)(hin + (size_t)old * HC))[q];
  float4 r; r.x = v.x*f; r.y = v.y*f; r.z = v.z*f; r.w = v.w*f;
  ((float4*)(hout + (size_t)i * HC))[q] = r;
}

__global__ void k_remap(const int* __restrict__ src, const int* __restrict__ dst,
                        const int* __restrict__ newid, int* __restrict__ ns,
                        int* __restrict__ nd, int E) {
  int e = blockIdx.x * blockDim.x + threadIdx.x;
  if (e >= E) return;
  int a = newid[src[e]], b = newid[dst[e]];
  if (a < 0 || b < 0) { ns[e] = -1; nd[e] = -1; }
  else                { ns[e] = a;  nd[e] = b;  }
}

// ---------------- attentional aggregation ----------------
__global__ void k_gate(const float* __restrict__ h, const float* __restrict__ gw,
                       const float* __restrict__ gb, float* __restrict__ gate, int Nn) {
  int n = blockIdx.x * blockDim.x + threadIdx.x;
  if (n >= Nn) return;
  const float4* hp = (const float4*)(h + (size_t)n * HC);
  const float4* wp = (const float4*)gw;
  float dp = 0.f;
  for (int q = 0; q < HC / 4; ++q) {
    float4 a = hp[q], b = wp[q];
    dp += a.x*b.x + a.y*b.y + a.z*b.z + a.w*b.w;
  }
  gate[n] = dp + gb[0];
}

// block per graph, 256 threads; npg <= 1024
__global__ __launch_bounds__(256)
void k_attpool(const float* __restrict__ h, const float* __restrict__ gate,
               float* __restrict__ out, int npg, int accumulate) {
  __shared__ float red[256];
  __shared__ float wsh[1024];
  const int b = blockIdx.x, t = threadIdx.x;
  const float* gg = gate + b * npg;
  float m = -3.4e38f;
  for (int i = t; i < npg; i += 256) m = fmaxf(m, gg[i]);
  red[t] = m; __syncthreads();
  for (int s = 128; s > 0; s >>= 1) { if (t < s) red[t] = fmaxf(red[t], red[t + s]); __syncthreads(); }
  const float gmax = red[0]; __syncthreads();
  float sacc = 0.f;
  for (int i = t; i < npg; i += 256) { float e = __expf(gg[i] - gmax); wsh[i] = e; sacc += e; }
  red[t] = sacc; __syncthreads();
  for (int s = 128; s > 0; s >>= 1) { if (t < s) red[t] += red[t + s]; __syncthreads(); }
  const float inv = 1.f / red[0]; __syncthreads();
  const float* hb = h + (size_t)b * npg * HC;
  float acc = 0.f;
  for (int i = 0; i < npg; ++i) acc += wsh[i] * hb[(size_t)i * HC + t];
  float v = acc * inv;
  if (accumulate) out[b * HC + t] += v; else out[b * HC + t] = v;
}

// ---------------- host orchestration ----------------
static inline int cdiv(int a, int b) { return (a + b - 1) / b; }

extern "C" void kernel_launch(void* const* d_in, const int* in_sizes, int n_in,
                              void* d_out, int out_size, void* d_ws, size_t ws_size,
                              hipStream_t stream) {
  (void)in_sizes; (void)n_in; (void)out_size; (void)ws_size;
  const float* x       = (const float*)d_in[0];
  const int*   ei      = (const int*)d_in[1];
  const int*   src0    = ei;
  const int*   dst0    = ei + ETOT;
  const float* lin0_w  = (const float*)d_in[2];
  const float* lin0_b  = (const float*)d_in[3];
  const float* gat0_W  = (const float*)d_in[4];
  const float* gat0_as = (const float*)d_in[5];
  const float* gat0_ad = (const float*)d_in[6];
  const float* gat0_b  = (const float*)d_in[7];
  const float* pool0_w = (const float*)d_in[8];
  const float* gat1_W  = (const float*)d_in[9];
  const float* gat1_as = (const float*)d_in[10];
  const float* gat1_ad = (const float*)d_in[11];
  const float* gat1_b  = (const float*)d_in[12];
  const float* pool1_w = (const float*)d_in[13];
  const float* gate_w  = (const float*)d_in[14];
  const float* gate_b  = (const float*)d_in[15];
  float* out = (float*)d_out;

  // ---- carve workspace ----
  char* base = (char*)d_ws;
  size_t off = 0;
  auto carve = [&](size_t bytes) -> char* {
    char* p = base + off;
    off += (bytes + 255) & ~(size_t)255;
    return p;
  };
  float*  H     = (float*)carve((size_t)NTOT * HC * 4);    // h0 / gat-acc / h1 / h2
  float*  G     = (float*)carve((size_t)NTOT * HC * 4);    // g0 / g1
  __bf16* ABF   = (__bf16*)carve((size_t)NTOT * HC * 2);   // bf16 A for all GEMMs
  __bf16* BBF   = (__bf16*)carve((size_t)HC * HC * 2);     // bf16 Bt (col-major) per GEMM
  float*  AS    = (float*)carve((size_t)NTOT * NHEAD * 4);
  float*  AD    = (float*)carve((size_t)NTOT * NHEAD * 4);
  float*  LOGIT = (float*)carve((size_t)(ETOT + NTOT) * NHEAD * 4);
  float*  MX    = (float*)carve((size_t)NTOT * NHEAD * 4);
  float*  DEN   = (float*)carve((size_t)NTOT * NHEAD * 4);
  float*  SCORE = (float*)carve((size_t)NTOT * 4);
  float*  TFAC  = (float*)carve((size_t)(NTOT / 2) * 4);
  int*    PERM  = (int*)  carve((size_t)(NTOT / 2) * 4);
  int*    NEWID = (int*)  carve((size_t)NTOT * 4);
  float*  HP1   = (float*)carve((size_t)(NTOT / 2) * HC * 4);  // 32768 x 256
  int*    SRC1  = (int*)  carve((size_t)ETOT * 4);
  int*    DST1  = (int*)  carve((size_t)ETOT * 4);
  float*  HP2   = (float*)carve((size_t)(NTOT / 4) * HC * 4);  // 16384 x 256
  float*  GATE  = (float*)carve((size_t)(NTOT / 2) * 4);

  const int N1 = NTOT / 2;   // 32768 nodes after pool0
  const int N2 = NTOT / 4;   // 16384 nodes after pool1

  // 1) h0 = relu(x @ lin0_w + lin0_b)           [WMMA bf16, K padded 129->160]
  k_cvt_a_bf16<<<NTOT, KPAD0, 0, stream>>>(x, ABF, FIN, KPAD0);
  k_cvt_b_bf16<<<HC, KPAD0, 0, stream>>>(lin0_w, BBF, FIN, KPAD0);
  k_gemm_bf16<<<NTOT / 16, 256, 0, stream>>>(ABF, BBF, lin0_b, H, KPAD0, 1);
  // 2) g0 = h0 @ gat0_W                          [WMMA]
  k_cvt_a_bf16<<<NTOT, HC, 0, stream>>>(H, ABF, HC, HC);
  k_cvt_b_bf16<<<HC, HC, 0, stream>>>(gat0_W, BBF, HC, HC);
  k_gemm_bf16<<<NTOT / 16, 256, 0, stream>>>(ABF, BBF, nullptr, G, HC, 0);
  // 3) attention dots
  k_att<<<cdiv(NTOT * NHEAD, 256), 256, 0, stream>>>(G, gat0_as, gat0_ad, AS, AD, NTOT);
  // 4) init accumulators (H reused as GAT output accumulator)
  k_fill_f32<<<cdiv(NTOT * NHEAD, 256), 256, 0, stream>>>(MX, -1e30f, NTOT * NHEAD);
  k_fill_f32<<<cdiv(NTOT * NHEAD, 256), 256, 0, stream>>>(DEN, 0.f, NTOT * NHEAD);
  k_fill_f32<<<cdiv(NTOT * HC, 256), 256, 0, stream>>>(H, 0.f, NTOT * HC);
  // 5-7) segment softmax + scatter (incl. self loops)
  k_edge_logit<<<cdiv(ETOT + NTOT, 256), 256, 0, stream>>>(src0, dst0, AS, AD, LOGIT, MX, ETOT, NTOT);
  k_edge_expsum<<<cdiv(ETOT + NTOT, 256), 256, 0, stream>>>(dst0, MX, LOGIT, DEN, ETOT, NTOT);
  k_edge_scatter<<<cdiv((ETOT + NTOT) * NHEAD, 256), 256, 0, stream>>>(src0, dst0, LOGIT, DEN, G, H, ETOT, NTOT);
  // 8) h1 = elu(acc + gat0_b)
  k_bias_elu<<<cdiv(NTOT * HC, 256), 256, 0, stream>>>(H, gat0_b, NTOT * HC);
  // 9-13) TopK pool 0: 2048 -> 1024 per graph
  k_score<<<cdiv(NTOT, 256), 256, 0, stream>>>(H, pool0_w, SCORE, NTOT);
  k_fill_i32<<<cdiv(NTOT, 256), 256, 0, stream>>>(NEWID, -1, NTOT);
  k_topk<<<NB, NPG0 / 2, 0, stream>>>(SCORE, TFAC, PERM, NEWID, NPG0);
  k_pool_gather<<<cdiv(N1 * (HC / 4), 256), 256, 0, stream>>>(H, PERM, TFAC, HP1, N1);
  k_remap<<<cdiv(ETOT, 256), 256, 0, stream>>>(src0, dst0, NEWID, SRC1, DST1, ETOT);
  // 14-15) out = attpool(HP1)
  k_gate<<<cdiv(N1, 256), 256, 0, stream>>>(HP1, gate_w, gate_b, GATE, N1);
  k_attpool<<<NB, 256, 0, stream>>>(HP1, GATE, out, NPG0 / 2, 0);
  // 16) g1 = HP1 @ gat1_W                        [WMMA]
  k_cvt_a_bf16<<<N1, HC, 0, stream>>>(HP1, ABF, HC, HC);
  k_cvt_b_bf16<<<HC, HC, 0, stream>>>(gat1_W, BBF, HC, HC);
  k_gemm_bf16<<<N1 / 16, 256, 0, stream>>>(ABF, BBF, nullptr, G, HC, 0);
  // 17) attention dots (layer 1)
  k_att<<<cdiv(N1 * NHEAD, 256), 256, 0, stream>>>(G, gat1_as, gat1_ad, AS, AD, N1);
  // 18) init
  k_fill_f32<<<cdiv(N1 * NHEAD, 256), 256, 0, stream>>>(MX, -1e30f, N1 * NHEAD);
  k_fill_f32<<<cdiv(N1 * NHEAD, 256), 256, 0, stream>>>(DEN, 0.f, N1 * NHEAD);
  k_fill_f32<<<cdiv(N1 * HC, 256), 256, 0, stream>>>(H, 0.f, N1 * HC);
  // 19-21) segment softmax + scatter on pooled graph (filtered edges, -1 = invalid)
  k_edge_logit<<<cdiv(ETOT + N1, 256), 256, 0, stream>>>(SRC1, DST1, AS, AD, LOGIT, MX, ETOT, N1);
  k_edge_expsum<<<cdiv(ETOT + N1, 256), 256, 0, stream>>>(DST1, MX, LOGIT, DEN, ETOT, N1);
  k_edge_scatter<<<cdiv((ETOT + N1) * NHEAD, 256), 256, 0, stream>>>(SRC1, DST1, LOGIT, DEN, G, H, ETOT, N1);
  // 22) h2 = elu(acc + gat1_b)
  k_bias_elu<<<cdiv(N1 * HC, 256), 256, 0, stream>>>(H, gat1_b, N1 * HC);
  // 23-25) TopK pool 1: 1024 -> 512 per graph (no edge remap needed afterwards)
  k_score<<<cdiv(N1, 256), 256, 0, stream>>>(H, pool1_w, SCORE, N1);
  k_topk<<<NB, (NPG0 / 2) / 2, 0, stream>>>(SCORE, TFAC, PERM, NEWID, NPG0 / 2);
  k_pool_gather<<<cdiv(N2 * (HC / 4), 256), 256, 0, stream>>>(H, PERM, TFAC, HP2, N2);
  // 26-27) out += attpool(HP2)
  k_gate<<<cdiv(N2, 256), 256, 0, stream>>>(HP2, gate_w, gate_b, GATE, N2);
  k_attpool<<<NB, 256, 0, stream>>>(HP2, GATE, out, NPG0 / 4, 1);
}
